// TemporalGraphConvNet_19576460935366
// MI455X (gfx1250) — matrix operand order repaired and checked by reading, not compiled, not run on hardware
//
#include <hip/hip_runtime.h>
#include <hip/hip_bf16.h>

// ---- problem constants -------------------------------------------------
#define BATCH 256
#define NNODE 400
#define TDIM  128
#define HDIM  256
#define OUTD  800          // DIMENSIONS * VOCAB = 400*2
#define NPAD  416          // NNODE padded to multiple of 32 for WMMA K-loop
#define NT    25           // NNODE / 16 row tiles
#define KSTEPS_AN (NPAD / 32)   // 13
#define KSTEPS_H  (HDIM / 32)   // 8
#define LOG_MAX_POS 9.210340371976184f

typedef __attribute__((ext_vector_type(16))) __bf16 v16bf;
typedef __attribute__((ext_vector_type(8)))  __bf16 v8bf;
typedef __attribute__((ext_vector_type(8)))  float  v8f;

// ---- kernel 0: zero the pooled accumulator -----------------------------
__global__ __launch_bounds__(256) void k_zero(float* __restrict__ p, int n) {
  int t = blockIdx.x * 256 + threadIdx.x;
  if (t < n) p[t] = 0.f;
}

// ---- kernel 1: column degree sums + GCN normalization ------------------
// degree[b,j] = sum_i adj[b,i,j]; dis[b,j] = rsqrt(#nonzero + 1)
__global__ __launch_bounds__(256) void k_degree(const float* __restrict__ adj,
                                                float* __restrict__ degree,
                                                float* __restrict__ dis) {
  int t = blockIdx.x * 256 + threadIdx.x;          // 0 .. B*N-1 (exact)
  int b = t / NNODE;
  int j = t - b * NNODE;
  const float* base = adj + (size_t)b * NNODE * NNODE + j;
  float ds = 0.f, cnt = 0.f;
  for (int i = 0; i < NNODE; ++i) {
    float v = base[(size_t)i * NNODE];             // coalesced over j
    ds  += v;
    cnt += (v != 0.f) ? 1.f : 0.f;
  }
  degree[t] = ds;
  dis[t]    = rsqrtf(cnt + 1.f);
}

// ---- kernel 2: sinusoidal time embedding folded into W1 ----------------
// TW[b,h] = sum_k temb[b,k] * W1[1+k, h]
__global__ __launch_bounds__(256) void k_temb_tw(const float* __restrict__ times,
                                                 const float* __restrict__ W1,
                                                 float* __restrict__ TW) {
  __shared__ float temb[TDIM];
  const int b = blockIdx.x;
  const int tid = threadIdx.x;                      // = h
  if (tid < TDIM) {
    int k = tid & 63;                               // half = 64
    float f = expf(-LOG_MAX_POS * (float)k / 63.0f);
    float arg = times[b] * f;
    temb[tid] = (tid < 64) ? sinf(arg) : cosf(arg);
  }
  __syncthreads();
  float acc = 0.f;
  for (int k = 0; k < TDIM; ++k)
    acc += temb[k] * W1[(size_t)(1 + k) * HDIM + tid];
  TW[b * HDIM + tid] = acc;
}

// ---- kernel 3: pack W2 transposed as bf16 (K contiguous per h row) -----
__global__ __launch_bounds__(256) void k_w2t(const float* __restrict__ W2,
                                             __bf16* __restrict__ W2T) {
  int t = blockIdx.x * 256 + threadIdx.x;           // HDIM*HDIM exact
  int h = t >> 8;
  int k = t & 255;
  W2T[(size_t)h * HDIM + k] = (__bf16)W2[(size_t)k * HDIM + h];
}

// ---- kernel 4: s0 = An*1, s1 = An*degree  (wave-per-row reductions) ----
__global__ __launch_bounds__(256) void k_s01(const float* __restrict__ adj,
                                             const float* __restrict__ degree,
                                             const float* __restrict__ dis,
                                             float* __restrict__ s0,
                                             float* __restrict__ s1) {
  int lane = threadIdx.x & 31;
  int wave = threadIdx.x >> 5;
  int rowg = blockIdx.x * 8 + wave;                 // 0 .. B*N-1 (exact)
  int b = rowg / NNODE;
  int i = rowg - b * NNODE;
  const float* arow = adj + ((size_t)(b * NNODE + i)) * NNODE;
  const float* dg = degree + b * NNODE;
  const float* dd = dis + b * NNODE;
  float a0 = 0.f, a1 = 0.f;
  for (int j = lane; j < NNODE; j += 32) {
    float v = arow[j];
    float w = ((v != 0.f) ? 1.f : 0.f) * dd[j];
    a0 += w;
    a1 += w * dg[j];
  }
  for (int m = 16; m > 0; m >>= 1) {
    a0 += __shfl_xor(a0, m, 32);
    a1 += __shfl_xor(a1, m, 32);
  }
  if (lane == 0) {
    float di = dd[i];                               // self loop: As[i,i] = 1
    s0[rowg] = di * (a0 + di);
    s1[rowg] = di * (a1 + di * dg[i]);
  }
}

// ---- device helper: A fragment from 16x32 LDS tile ---------------------
__device__ __forceinline__ v16bf load_afrag(const __bf16* tile, int row, int koff) {
  v8bf alo = *reinterpret_cast<const v8bf*>(&tile[row * 32 + 2 * koff]);
  v8bf ahi = *reinterpret_cast<const v8bf*>(&tile[row * 32 + 2 * koff + 8]);
  return __builtin_shufflevector(alo, ahi, 0,1,2,3,4,5,6,7,8,9,10,11,12,13,14,15);
}

// ---- kernel 5: GT = (relu(h1) @ W2)^T via bf16 WMMA --------------------
// h1 generated on the fly: h1[i,k] = relu(s1[i]*W1[0,k] + s0[i]*TW[b,k] + b1[k])
// block = 256 threads = 8 waves; wave w owns h-tiles {2w,2w+1} (16x32 strip);
// A tile double-buffered in LDS; 2 WMMAs per K-step reuse one A fragment.
__global__ __launch_bounds__(256) void k_h1_gemm(const float* __restrict__ s0,
                                                 const float* __restrict__ s1,
                                                 const float* __restrict__ TW,
                                                 const float* __restrict__ W1,
                                                 const float* __restrict__ b1,
                                                 const __bf16* __restrict__ W2T,
                                                 __bf16* __restrict__ GT) {
  __shared__ __bf16 tileA[2][16 * 32];
  const int b     = blockIdx.x / NT;
  const int ibase = (blockIdx.x % NT) * 16;
  const int tid   = threadIdx.x;
  const int lane  = tid & 31;
  const int wave  = tid >> 5;
  const int hbase = wave * 32;
  const int row   = lane & 15;                      // A row / D column
  const int sel   = lane >> 4;                      // lane half select
  const int koff  = sel * 8;
  const float s1i0 = s1[b * NNODE + ibase + (tid >> 5)];       // fill elem 0 row
  const float s0i0 = s0[b * NNODE + ibase + (tid >> 5)];
  const float s1i1 = s1[b * NNODE + ibase + 8 + (tid >> 5)];   // fill elem 1 row
  const float s0i1 = s0[b * NNODE + ibase + 8 + (tid >> 5)];

  // cooperative fill: 512 elems, 256 threads -> elems tid (rows 0-7 part)
  // laid out as e = tid (ii=tid>>5, kk=tid&31) and e+256 (ii=8+(tid>>5))
  auto fill = [&](int buf, int kbase) {
    int ii = tid >> 5;
    int kk = tid & 31;
    int k  = kbase + kk;
    float w1k = W1[k];
    float twk = TW[b * HDIM + k];
    float b1k = b1[k];
    float v0 = s1i0 * w1k + s0i0 * twk + b1k;
    float v1 = s1i1 * w1k + s0i1 * twk + b1k;
    tileA[buf][ii * 32 + kk]       = (__bf16)fmaxf(v0, 0.f);
    tileA[buf][(ii + 8) * 32 + kk] = (__bf16)fmaxf(v1, 0.f);
  };

  v8f c0 = {}, c1 = {};
  fill(0, 0);
  __syncthreads();
  for (int s = 0; s < KSTEPS_H; ++s) {
    int cur = s & 1;
    if (s + 1 < KSTEPS_H) fill(cur ^ 1, (s + 1) * 32);
    int kbase = s * 32;
    v16bf a = load_afrag(tileA[cur], row, koff);
    const v16bf bm0 = *reinterpret_cast<const v16bf*>(
        W2T + (size_t)(hbase + row) * HDIM + kbase + sel * 16);
    const v16bf bm1 = *reinterpret_cast<const v16bf*>(
        W2T + (size_t)(hbase + 16 + row) * HDIM + kbase + sel * 16);
    c0 = __builtin_amdgcn_wmma_f32_16x16x32_bf16(false, a, false, bm0,
                                                 (short)0, c0, false, false);
    c1 = __builtin_amdgcn_wmma_f32_16x16x32_bf16(false, a, false, bm1,
                                                 (short)0, c1, false, false);
    __syncthreads();
  }
  // store transposed: GT[b][h][i], lane holds 8 consecutive i per tile
  v8bf o0, o1;
#pragma unroll
  for (int r = 0; r < 8; ++r) { o0[r] = (__bf16)c0[r]; o1[r] = (__bf16)c1[r]; }
  *reinterpret_cast<v8bf*>(GT + ((size_t)(b * HDIM + hbase + row)) * NPAD
                              + ibase + sel * 8) = o0;
  *reinterpret_cast<v8bf*>(GT + ((size_t)(b * HDIM + hbase + 16 + row)) * NPAD
                              + ibase + sel * 8) = o1;
}

// ---- kernel 6: An @ GT^T, fused bias+relu+mean-pool --------------------
// An tile built in LDS from adj on the fly (double-buffered);
// wave owns a 16x32 output strip; pooled += mean_i relu(. + b2)
__global__ __launch_bounds__(256) void k_an_gemm_pool(const float* __restrict__ adj,
                                                      const float* __restrict__ dis,
                                                      const __bf16* __restrict__ GT,
                                                      const float* __restrict__ b2,
                                                      float* __restrict__ pooled) {
  __shared__ __bf16 tileA[2][16 * 32];
  const int b     = blockIdx.x / NT;
  const int ibase = (blockIdx.x % NT) * 16;
  const int tid   = threadIdx.x;
  const int lane  = tid & 31;
  const int wave  = tid >> 5;
  const int hbase = wave * 32;
  const int row   = lane & 15;
  const int sel   = lane >> 4;
  const int koff  = sel * 8;
  const float* disB = dis + b * NNODE;
  const int fi0 = ibase + (tid >> 5);               // fill rows
  const int fi1 = ibase + 8 + (tid >> 5);
  const float di0 = disB[fi0];
  const float di1 = disB[fi1];

  auto fill = [&](int buf, int jbase) {
    int kk = tid & 31;
    int j  = jbase + kk;
    float an0 = 0.f, an1 = 0.f;
    if (j < NNODE) {
      float dj = disB[j];
      float a0 = adj[((size_t)(b * NNODE + fi0)) * NNODE + j];
      float a1 = adj[((size_t)(b * NNODE + fi1)) * NNODE + j];
      float as0 = ((a0 != 0.f) ? 1.f : 0.f) + ((fi0 == j) ? 1.f : 0.f);
      float as1 = ((a1 != 0.f) ? 1.f : 0.f) + ((fi1 == j) ? 1.f : 0.f);
      an0 = di0 * as0 * dj;
      an1 = di1 * as1 * dj;
    }
    int ii = tid >> 5;
    tileA[buf][ii * 32 + kk]       = (__bf16)an0;
    tileA[buf][(ii + 8) * 32 + kk] = (__bf16)an1;
  };

  v8f c0 = {}, c1 = {};
  fill(0, 0);
  __syncthreads();
  for (int s = 0; s < KSTEPS_AN; ++s) {
    int cur = s & 1;
    if (s + 1 < KSTEPS_AN) fill(cur ^ 1, (s + 1) * 32);
    int jbase = s * 32;
    v16bf a = load_afrag(tileA[cur], row, koff);
    const v16bf bm0 = *reinterpret_cast<const v16bf*>(
        GT + ((size_t)(b * HDIM + hbase + row)) * NPAD + jbase + sel * 16);
    const v16bf bm1 = *reinterpret_cast<const v16bf*>(
        GT + ((size_t)(b * HDIM + hbase + 16 + row)) * NPAD + jbase + sel * 16);
    c0 = __builtin_amdgcn_wmma_f32_16x16x32_bf16(false, a, false, bm0,
                                                 (short)0, c0, false, false);
    c1 = __builtin_amdgcn_wmma_f32_16x16x32_bf16(false, a, false, bm1,
                                                 (short)0, c1, false, false);
    __syncthreads();
  }
  // epilogue: bias + relu + mean pool (h2 never materialized)
  float bias0 = b2[hbase + row];
  float bias1 = b2[hbase + 16 + row];
  float p0 = 0.f, p1 = 0.f;
#pragma unroll
  for (int r = 0; r < 8; ++r) {
    p0 += fmaxf(c0[r] + bias0, 0.f);
    p1 += fmaxf(c1[r] + bias1, 0.f);
  }
  p0 *= (1.0f / (float)NNODE);
  p1 *= (1.0f / (float)NNODE);
  p0 += __shfl_xor(p0, 16, 32);                     // combine M 0-7 with 8-15
  p1 += __shfl_xor(p1, 16, 32);
  if (lane < 16) {
    atomicAdd(&pooled[b * HDIM + hbase + lane], p0);
    atomicAdd(&pooled[b * HDIM + hbase + 16 + lane], p1);
  }
}

// ---- kernel 7: logits = pooled @ Wlin + blin ---------------------------
__global__ __launch_bounds__(256) void k_final(const float* __restrict__ pooled,
                                               const float* __restrict__ Wlin,
                                               const float* __restrict__ blin,
                                               float* __restrict__ out) {
  int t = blockIdx.x * 256 + threadIdx.x;
  if (t >= BATCH * OUTD) return;
  int b = t / OUTD;
  int o = t - b * OUTD;
  float acc = blin[o];
  const float* p = pooled + b * HDIM;
  for (int k = 0; k < HDIM; ++k) acc += p[k] * Wlin[(size_t)k * OUTD + o];
  out[t] = acc;
}

// ---- launch ------------------------------------------------------------
extern "C" void kernel_launch(void* const* d_in, const int* in_sizes, int n_in,
                              void* d_out, int out_size, void* d_ws, size_t ws_size,
                              hipStream_t stream) {
  (void)in_sizes; (void)n_in; (void)out_size; (void)ws_size;
  const float* adj   = (const float*)d_in[0];
  const float* times = (const float*)d_in[1];
  const float* W1    = (const float*)d_in[2];
  const float* b1    = (const float*)d_in[3];
  const float* W2    = (const float*)d_in[4];
  const float* b2    = (const float*)d_in[5];
  const float* Wlin  = (const float*)d_in[6];
  const float* blin  = (const float*)d_in[7];
  float* out = (float*)d_out;

  // workspace layout (256B-aligned offsets)
  char* ws = (char*)d_ws;
  size_t off = 0;
  float* pooled = (float*)(ws + off); off += (size_t)BATCH * HDIM * 4;          // 256 KB
  float* degree = (float*)(ws + off); off += (size_t)BATCH * NNODE * 4;         // 400 KB
  float* dis    = (float*)(ws + off); off += (size_t)BATCH * NNODE * 4;
  float* s0     = (float*)(ws + off); off += (size_t)BATCH * NNODE * 4;
  float* s1     = (float*)(ws + off); off += (size_t)BATCH * NNODE * 4;
  float* TW     = (float*)(ws + off); off += (size_t)BATCH * HDIM * 4;
  __bf16* W2T   = (__bf16*)(ws + off); off += (size_t)HDIM * HDIM * 2;
  off = (off + 255) & ~(size_t)255;
  __bf16* GT    = (__bf16*)(ws + off); off += (size_t)BATCH * HDIM * NPAD * 2;  // ~54.5 MB

  k_zero<<<(BATCH * HDIM + 255) / 256, 256, 0, stream>>>(pooled, BATCH * HDIM);
  k_degree<<<(BATCH * NNODE) / 256, 256, 0, stream>>>(adj, degree, dis);
  k_temb_tw<<<BATCH, 256, 0, stream>>>(times, W1, TW);
  k_w2t<<<(HDIM * HDIM) / 256, 256, 0, stream>>>(W2, W2T);
  k_s01<<<(BATCH * NNODE) / 8, 256, 0, stream>>>(adj, degree, dis, s0, s1);
  k_h1_gemm<<<BATCH * NT, 256, 0, stream>>>(s0, s1, TW, W1, b1, W2T, GT);
  k_an_gemm_pool<<<BATCH * NT, 256, 0, stream>>>(adj, dis, GT, b2, pooled);
  k_final<<<(BATCH * OUTD + 255) / 256, 256, 0, stream>>>(pooled, Wlin, blin, out);
}
